// JointSlotFusion_33698313404761
// MI455X (gfx1250) — matrix operand order repaired and checked by reading, not compiled, not run on hardware
//
#include <hip/hip_runtime.h>
#include <hip/hip_bf16.h>

typedef __attribute__((ext_vector_type(16))) _Float16 v16h;
typedef __attribute__((ext_vector_type(8)))  _Float16 v8h;
typedef __attribute__((ext_vector_type(8)))  float    v8f;

#define C_IN 256
#define HW_  1024
#define DD   64

// ---------------------------------------------------------------------------
// Kernel P: one-time conversion of GEMM weights fp32 -> f16 into workspace.
// Layouts preserved: conv w [64][256], Wk/Wv [64][64].
// ---------------------------------------------------------------------------
__global__ __launch_bounds__(256) void prep_weights(
    const float* __restrict__ cw1, const float* __restrict__ cw2,
    const float* __restrict__ Wk,  const float* __restrict__ Wv,
    _Float16* __restrict__ wh)   // [w1h 16384][w2h 16384][wkh 4096][wvh 4096]
{
    int i = blockIdx.x * 256 + threadIdx.x;
    if (i >= 40960) return;
    float v;
    if      (i < 16384) v = cw1[i];
    else if (i < 32768) v = cw2[i - 16384];
    else if (i < 36864) v = Wk [i - 32768];
    else                v = Wv [i - 36864];
    wh[i] = (_Float16)v;
}

// ---------------------------------------------------------------------------
// Kernel A: conv1x1 (both sources) + bias + LN(norm) + LN(ni) + K/V projection
// One wave handles a 16-token x 64-channel tile via f16 WMMA (f32 accumulate).
// ---------------------------------------------------------------------------
__global__ __launch_bounds__(256) void fused_tokens_kv(
    const float* __restrict__ x1, const float* __restrict__ x2,
    const float* __restrict__ cb1, const float* __restrict__ cb2,
    const float* __restrict__ nw,  const float* __restrict__ nb,
    const float* __restrict__ iw,  const float* __restrict__ ib,
    const _Float16* __restrict__ wh,     // packed f16 weights (prep_weights)
    _Float16* __restrict__ kbuf, _Float16* __restrict__ vbuf)
{
    __shared__ _Float16 tlds[8][16][DD];   // per-wave 16x64 f16 transpose tile

    const int lane = threadIdx.x & 31;
    const int wv   = threadIdx.x >> 5;
    const int hi   = lane >> 4;            // half-wave select
    const int nlo  = lane & 15;

    const int  tile   = blockIdx.x * 8 + wv;        // global 16-token tile id
    const long tokAbs = (long)tile * 16;            // absolute token index
    const int  b      = (int)(tokAbs >> 11);        // / 2048
    const int  nIB    = (int)(tokAbs & 2047);       // token-in-batch
    const int  which  = nIB >> 10;                  // 0 -> x1, 1 -> x2
    const int  hw0    = nIB & 1023;

    const float*    xs = which ? x2 : x1;
    const _Float16* ws = wh + (which ? 16384 : 0);  // conv weights, f16
    const float*    bs = which ? cb2 : cb1;
    const _Float16* wkh = wh + 32768;
    const _Float16* wvh = wh + 36864;
    const float*    xb = xs + (long)b * C_IN * HW_;

    // ---- conv GEMM: [16 x 256] x [256 x 64] in 8 k-steps of 32 -------------
    v8f acc[4] = {};
    for (int kk = 0; kk < 8; ++kk) {
        const int k0 = kk * 32;
        v16h A;
        #pragma unroll
        for (int j = 0; j < 16; ++j) {
            // A-layout: lanes0-15 K={0..15}, lanes16-31 K={16..31}
            int koff = j + hi * 16;
            A[j] = (_Float16)xb[(long)(k0 + koff) * HW_ + hw0 + nlo];
        }
        #pragma unroll
        for (int nt = 0; nt < 4; ++nt) {
            // B[k][n] = W[n][k]: 16 contiguous f16 per lane (32 B)
            v16h Bf = *(const v16h*)(ws + (nt * 16 + nlo) * C_IN + k0 + hi * 16);
            acc[nt] = __builtin_amdgcn_wmma_f32_16x16x32_f16(
                false, A, false, Bf, (short)0, acc[nt], false, false);
        }
    }

    // ---- bias -------------------------------------------------------------
    #pragma unroll
    for (int nt = 0; nt < 4; ++nt) {
        float bia = bs[nt * 16 + nlo];
        #pragma unroll
        for (int r = 0; r < 8; ++r) acc[nt][r] += bia;
    }

    // ---- two layernorms over D=64 (row = token, striped across 16 lanes) --
    for (int pass = 0; pass < 2; ++pass) {
        const float* gw = pass ? iw : nw;
        const float* gb = pass ? ib : nb;
        float g[4], gB[4];
        #pragma unroll
        for (int nt = 0; nt < 4; ++nt) {
            g[nt]  = gw[nt * 16 + nlo];
            gB[nt] = gb[nt * 16 + nlo];
        }
        #pragma unroll
        for (int r = 0; r < 8; ++r) {
            float s = 0.f, s2 = 0.f;
            #pragma unroll
            for (int nt = 0; nt < 4; ++nt) {
                float v = acc[nt][r]; s += v; s2 += v * v;
            }
            #pragma unroll
            for (int off = 8; off >= 1; off >>= 1) {   // stays within half-wave
                s  += __shfl_xor(s,  off, 32);
                s2 += __shfl_xor(s2, off, 32);
            }
            float mu  = s  * (1.f / 64.f);
            float var = s2 * (1.f / 64.f) - mu * mu;
            float rs  = rsqrtf(var + 1e-5f);
            #pragma unroll
            for (int nt = 0; nt < 4; ++nt)
                acc[nt][r] = (acc[nt][r] - mu) * rs * g[nt] + gB[nt];
        }
    }

    // ---- transpose C-layout -> A-layout through LDS (f16) ------------------
    #pragma unroll
    for (int r = 0; r < 8; ++r)
        #pragma unroll
        for (int nt = 0; nt < 4; ++nt)
            tlds[wv][r + hi * 8][nt * 16 + nlo] = (_Float16)acc[nt][r];
    __syncthreads();

    // ---- K/V projections: [16 x 64] x [64 x 64], 2 k-steps -----------------
    v8f ka[4] = {}, va[4] = {};
    #pragma unroll
    for (int ks = 0; ks < 2; ++ks) {
        const int k0 = ks * 32;
        v8h lo  = *(const v8h*)&tlds[wv][nlo][k0 + hi * 16];
        v8h hi8 = *(const v8h*)&tlds[wv][nlo][k0 + hi * 16 + 8];
        v16h A;
        #pragma unroll
        for (int j = 0; j < 8; ++j) { A[j] = lo[j]; A[j + 8] = hi8[j]; }
        #pragma unroll
        for (int nt = 0; nt < 4; ++nt) {
            v16h Bk = *(const v16h*)(wkh + (nt * 16 + nlo) * DD + k0 + hi * 16);
            v16h Bv = *(const v16h*)(wvh + (nt * 16 + nlo) * DD + k0 + hi * 16);
            ka[nt] = __builtin_amdgcn_wmma_f32_16x16x32_f16(
                false, A, false, Bk, (short)0, ka[nt], false, false);
            va[nt] = __builtin_amdgcn_wmma_f32_16x16x32_f16(
                false, A, false, Bv, (short)0, va[nt], false, false);
        }
    }

    // ---- store k, v as f16 --------------------------------------------------
    #pragma unroll
    for (int r = 0; r < 8; ++r) {
        #pragma unroll
        for (int nt = 0; nt < 4; ++nt) {
            long idx = (tokAbs + r + hi * 8) * DD + nt * 16 + nlo;
            kbuf[idx] = (_Float16)ka[nt][r];
            vbuf[idx] = (_Float16)va[nt][r];
        }
    }
}

// ---------------------------------------------------------------------------
// Kernel B: 3 slot-attention iterations + head. One workgroup (16 waves) per
// batch. Logits and updates GEMMs run on WMMA; GRU/MLP stay VALU (tiny).
// ---------------------------------------------------------------------------
__global__ __launch_bounds__(512) void slot_attention(
    const _Float16* __restrict__ kbuf, const _Float16* __restrict__ vbuf,
    const float* __restrict__ smu,  const float* __restrict__ lsig,
    const float* __restrict__ noise,
    const float* __restrict__ nsw,  const float* __restrict__ nsb,
    const float* __restrict__ nmw,  const float* __restrict__ nmb,
    const float* __restrict__ Wq,
    const float* __restrict__ wih,  const float* __restrict__ whh,
    const float* __restrict__ bih,  const float* __restrict__ bhh,
    const float* __restrict__ w1m,  const float* __restrict__ b1m,
    const float* __restrict__ w2m,  const float* __restrict__ b2m,
    const float* __restrict__ hwm,  const float* __restrict__ hbm,
    float* __restrict__ out)
{
    __shared__ _Float16 attnT[8][2048];           // 32 KB, [slot][token]
    __shared__ _Float16 qh[8][64];                // q as f16 for WMMA B-frags
    __shared__ float slots[8][64], sprev[8][64], sn[8][64];
    __shared__ float upd[8][64], gi[8][192], gh[8][192], h1[8][128];
    __shared__ float colsum[8], fused[64];

    const int tid  = threadIdx.x;
    const int lane = tid & 31;
    const int wv   = tid >> 5;                    // 0..15
    const int hi   = lane >> 4;
    const int nlo  = lane & 15;
    const int b    = blockIdx.x;
    const _Float16* kb = kbuf + (long)b * 2048 * 64;
    const _Float16* vb = vbuf + (long)b * 2048 * 64;

    { // slot init: mu + exp(log_sigma) * noise
        int s = tid >> 6, d = tid & 63;
        slots[s][d] = smu[d] + expf(lsig[d]) * noise[(long)b * 512 + tid];
    }
    __syncthreads();

    for (int iter = 0; iter < 3; ++iter) {
        { int s = tid >> 6, d = tid & 63; sprev[s][d] = slots[s][d]; }
        __syncthreads();
        if (tid < 8) {                         // sn = LN(slots, ns)
            float m = 0.f;
            for (int d = 0; d < 64; ++d) m += slots[tid][d];
            m *= (1.f / 64.f);
            float v = 0.f;
            for (int d = 0; d < 64; ++d) { float t = slots[tid][d] - m; v += t * t; }
            float rs = rsqrtf(v * (1.f / 64.f) + 1e-5f);
            for (int d = 0; d < 64; ++d)
                sn[tid][d] = (slots[tid][d] - m) * rs * nsw[d] + nsb[d];
        }
        __syncthreads();
        {                                      // q = (sn @ Wq^T) * D^-0.5, f16
            int s = tid >> 6, d = tid & 63;
            float a = 0.f;
            for (int j = 0; j < 64; ++j) a += sn[s][j] * Wq[d * 64 + j];
            qh[s][d] = (_Float16)(a * 0.125f);
        }
        __syncthreads();

        // ---- logits via WMMA + softmax over S, write attnT ------------------
        // Each wave handles 8 token tiles of 16.
        for (int i = 0; i < 8; ++i) {
            const int tok0 = (wv * 8 + i) * 16;
            v8f c = {};
            #pragma unroll
            for (int ks = 0; ks < 2; ++ks) {
                const int k0 = ks * 32;
                const _Float16* ar = kb + (long)(tok0 + nlo) * 64 + k0 + hi * 16;
                v8h alo = *(const v8h*)(ar);
                v8h ahi = *(const v8h*)(ar + 8);
                v16h A, Bq;
                #pragma unroll
                for (int j = 0; j < 8; ++j) { A[j] = alo[j]; A[j + 8] = ahi[j]; }
                if (nlo < 8) {
                    Bq = *(const v16h*)&qh[nlo][k0 + hi * 16];
                } else {
                    Bq = (v16h){};
                }
                c = __builtin_amdgcn_wmma_f32_16x16x32_f16(
                    false, A, false, Bq, (short)0, c, false, false);
            }
            #pragma unroll
            for (int r = 0; r < 8; ++r) {      // softmax across 8 slot lanes
                float l  = c[r];
                float mx = l;
                #pragma unroll
                for (int off = 4; off >= 1; off >>= 1)
                    mx = fmaxf(mx, __shfl_xor(mx, off, 32));
                float e   = expf(l - mx);
                float den = e;
                #pragma unroll
                for (int off = 4; off >= 1; off >>= 1)
                    den += __shfl_xor(den, off, 32);
                float a = e / den + 1e-8f;
                if (nlo < 8) attnT[nlo][tok0 + r + hi * 8] = (_Float16)a;
            }
        }
        __syncthreads();

        if (tid < 8) colsum[tid] = 0.f;
        { int s = tid >> 6, d = tid & 63; upd[s][d] = 0.f; }
        __syncthreads();

        // ---- colsum partials + updates WMMA ---------------------------------
        {   // colsum: 64 threads per slot, strided
            int s = tid >> 6, n0 = tid & 63;
            float p = 0.f;
            for (int n = n0; n < 2048; n += 64) p += (float)attnT[s][n];
            atomicAdd(&colsum[s], p);
        }
        {   // updates: each wave reduces its 128-token K-slice (4 k-steps)
            v8f u[4] = {};
            #pragma unroll
            for (int i = 0; i < 4; ++i) {
                const int kt = wv * 128 + i * 32;
                v16h A;
                if (nlo < 8) {
                    v8h alo = *(const v8h*)&attnT[nlo][kt + hi * 16];
                    v8h ahi = *(const v8h*)&attnT[nlo][kt + hi * 16 + 8];
                    #pragma unroll
                    for (int j = 0; j < 8; ++j) { A[j] = alo[j]; A[j + 8] = ahi[j]; }
                } else {
                    A = (v16h){};
                }
                #pragma unroll
                for (int nt = 0; nt < 4; ++nt) {
                    const _Float16* vr = vb + (long)(kt + hi * 16) * 64 + nt * 16 + nlo;
                    v16h B;
                    #pragma unroll
                    for (int j = 0; j < 16; ++j) B[j] = vr[(long)j * 64];
                    u[nt] = __builtin_amdgcn_wmma_f32_16x16x32_f16(
                        false, A, false, B, (short)0, u[nt], false, false);
                }
            }
            if (hi == 0) {                     // rows 0..7 = valid slots
                #pragma unroll
                for (int r = 0; r < 8; ++r)
                    #pragma unroll
                    for (int nt = 0; nt < 4; ++nt)
                        atomicAdd(&upd[r][nt * 16 + nlo], u[nt][r]);
            }
        }
        __syncthreads();
        { int s = tid >> 6, d = tid & 63; upd[s][d] /= colsum[s]; }
        __syncthreads();

        // ---- GRU ------------------------------------------------------------
        for (int i = tid; i < 1536; i += 512) {
            int s = i / 192, j = i % 192;
            float a = bih[j], h = bhh[j];
            for (int d = 0; d < 64; ++d) {
                a += upd[s][d]   * wih[j * 64 + d];
                h += sprev[s][d] * whh[j * 64 + d];
            }
            gi[s][j] = a; gh[s][j] = h;
        }
        __syncthreads();
        {                                      // gates (r, z, n order)
            int s = tid >> 6, d = tid & 63;
            float r  = 1.f / (1.f + expf(-(gi[s][d]       + gh[s][d])));
            float z  = 1.f / (1.f + expf(-(gi[s][64 + d]  + gh[s][64 + d])));
            float nn = tanhf(gi[s][128 + d] + r * gh[s][128 + d]);
            slots[s][d] = (1.f - z) * nn + z * sprev[s][d];
        }
        __syncthreads();
        if (tid < 8) {                         // LN(slots, nm) -> sn (reuse)
            float m = 0.f;
            for (int d = 0; d < 64; ++d) m += slots[tid][d];
            m *= (1.f / 64.f);
            float v = 0.f;
            for (int d = 0; d < 64; ++d) { float t = slots[tid][d] - m; v += t * t; }
            float rs = rsqrtf(v * (1.f / 64.f) + 1e-5f);
            for (int d = 0; d < 64; ++d)
                sn[tid][d] = (slots[tid][d] - m) * rs * nmw[d] + nmb[d];
        }
        __syncthreads();
        for (int i = tid; i < 1024; i += 512) { // MLP layer 1 + relu
            int s = i >> 7, j = i & 127;
            float a = b1m[j];
            for (int d = 0; d < 64; ++d) a += sn[s][d] * w1m[j * 64 + d];
            h1[s][j] = fmaxf(a, 0.f);
        }
        __syncthreads();
        {                                      // MLP layer 2 + residual
            int s = tid >> 6, d = tid & 63;
            float a = b2m[d];
            for (int j = 0; j < 128; ++j) a += h1[s][j] * w2m[d * 128 + j];
            slots[s][d] += a;
        }
        __syncthreads();
    }

    if (tid < 64) {                            // fused = mean over slots
        float a = 0.f;
        #pragma unroll
        for (int s = 0; s < 8; ++s) a += slots[s][tid];
        fused[tid] = a * 0.125f;
    }
    __syncthreads();
    if (tid < 15) {                            // classifier head
        float a = hbm[tid];
        for (int d = 0; d < 64; ++d) a += fused[d] * hwm[tid * 64 + d];
        out[(long)b * 15 + tid] = a;
    }
}

// ---------------------------------------------------------------------------
extern "C" void kernel_launch(void* const* d_in, const int* in_sizes, int n_in,
                              void* d_out, int out_size, void* d_ws, size_t ws_size,
                              hipStream_t stream) {
    const float* x1       = (const float*)d_in[0];
    const float* x2       = (const float*)d_in[1];
    const float* conv1_w  = (const float*)d_in[2];
    const float* conv1_b  = (const float*)d_in[3];
    const float* conv2_w  = (const float*)d_in[4];
    const float* conv2_b  = (const float*)d_in[5];
    const float* norm_w   = (const float*)d_in[6];
    const float* norm_b   = (const float*)d_in[7];
    const float* ni_w     = (const float*)d_in[8];
    const float* ni_b     = (const float*)d_in[9];
    const float* ns_w     = (const float*)d_in[10];
    const float* ns_b     = (const float*)d_in[11];
    const float* nm_w     = (const float*)d_in[12];
    const float* nm_b     = (const float*)d_in[13];
    const float* slots_mu = (const float*)d_in[14];
    const float* slots_ls = (const float*)d_in[15];
    const float* Wq       = (const float*)d_in[16];
    const float* Wk       = (const float*)d_in[17];
    const float* Wv       = (const float*)d_in[18];
    const float* gru_wih  = (const float*)d_in[19];
    const float* gru_whh  = (const float*)d_in[20];
    const float* gru_bih  = (const float*)d_in[21];
    const float* gru_bhh  = (const float*)d_in[22];
    const float* mlp_w1   = (const float*)d_in[23];
    const float* mlp_b1   = (const float*)d_in[24];
    const float* mlp_w2   = (const float*)d_in[25];
    const float* mlp_b2   = (const float*)d_in[26];
    const float* head_w   = (const float*)d_in[27];
    const float* head_b   = (const float*)d_in[28];
    const float* noise    = (const float*)d_in[29];

    _Float16* kbuf = (_Float16*)d_ws;
    _Float16* vbuf = kbuf + (size_t)128 * 2048 * 64;   // 32 MB each
    _Float16* wh   = vbuf + (size_t)128 * 2048 * 64;   // 80 KB f16 weights

    prep_weights<<<160, 256, 0, stream>>>(conv1_w, conv2_w, Wk, Wv, wh);

    fused_tokens_kv<<<2048, 256, 0, stream>>>(
        x1, x2, conv1_b, conv2_b,
        norm_w, norm_b, ni_w, ni_b, wh, kbuf, vbuf);

    slot_attention<<<128, 512, 0, stream>>>(
        kbuf, vbuf, slots_mu, slots_ls, noise,
        ns_w, ns_b, nm_w, nm_b, Wq,
        gru_wih, gru_whh, gru_bih, gru_bhh,
        mlp_w1, mlp_b1, mlp_w2, mlp_b2,
        head_w, head_b, (float*)d_out);
}
